// GRNN_CFC_78443282694880
// MI455X (gfx1250) — matrix operand order repaired and checked by reading, not compiled, not run on hardware
//
#include <hip/hip_runtime.h>
#include <hip/hip_bf16.h>
#include <math.h>

// CfC bidirectional scan for MI455X (gfx1250, wave32, WMMA).
// B=128, T=1000, D=256, H=256, BB=128.
// Grid: 16 blocks (= 2 directions x 8 batch-slices of 16), 256 threads (8 waves).
// Each wave pins its WMMA B-fragments (weights) in VGPRs for the entire scan.
// Accumulators are zero-initialized so the first WMMA encodes SRC2 = inline 0;
// biases are folded into the elementwise epilogues.

typedef __attribute__((ext_vector_type(16))) _Float16 v16h;
typedef __attribute__((ext_vector_type(8)))  _Float16 v8h;
typedef __attribute__((ext_vector_type(8)))  float    v8f;

#define CFC_T  1000
#define CFC_D  256
#define CFC_H  256
#define CFC_BB 128

// Padded LDS row strides (in halfs). 264*2B=528B and 136*2B=272B are both
// 16B-aligned and give bank offsets of 4 DWORD-banks per row -> the 16 rows of
// an A-fragment read hit all 64 banks conflict-free.
#define SX_STRIDE 264
#define SZ_STRIDE 136

__device__ __forceinline__ void wmma_acc(v16h a, v16h b, v8f& acc) {
    acc = __builtin_amdgcn_wmma_f32_16x16x32_f16(
        /*neg_a=*/false, a, /*neg_b=*/false, b,
        /*c_mod=*/(short)0, acc, /*reuse_a=*/false, /*reuse_b=*/false);
}

// Load a 16x32 f16 A-fragment from an LDS tile (row-major, given stride).
// Contiguous K order used here (B frags match): lane L holds row M=L&15; lanes 0-15 carry K {0..15},
// lanes 16-31 carry K {16..31}. Two contiguous 16B chunks per lane.
__device__ __forceinline__ v16h load_afrag(const _Float16* tile, int stride, int lane) {
    int m  = lane & 15;
    int kb = (lane >> 4) << 4;            // 0 or 16
    const _Float16* p = tile + m * stride + kb;
    v8h lo = *(const v8h*)(p);            // K = kb .. kb+7
    v8h hi = *(const v8h*)(p + 8);        // K = kb+8 .. kb+15
    return __builtin_shufflevector(lo, hi, 0,1,2,3,4,5,6,7,8,9,10,11,12,13,14,15);
}

__global__ __launch_bounds__(256, 1)
void cfc_scan_kernel(const float* __restrict__ x,
                     const float* __restrict__ Wb1, const float* __restrict__ bb1,
                     const float* __restrict__ W11, const float* __restrict__ b11,
                     const float* __restrict__ wt1, const float* __restrict__ A1,
                     const float* __restrict__ Wb2, const float* __restrict__ bb2,
                     const float* __restrict__ W12, const float* __restrict__ b12,
                     const float* __restrict__ wt2, const float* __restrict__ A2,
                     float* __restrict__ out) {
    __shared__ __align__(16) _Float16 sX[2][16][SX_STRIDE]; // x_t slice, double buffered
    __shared__ __align__(16) _Float16 sH[16][SX_STRIDE];    // hidden state (f16 copy)
    __shared__ __align__(16) _Float16 sZ[16][SZ_STRIDE];    // backbone activations

    const int tid   = threadIdx.x;
    const int wave  = tid >> 5;          // 0..7
    const int lane  = tid & 31;
    const int dir   = blockIdx.x >> 3;   // 0 = forward, 1 = reversed input
    const int slice = blockIdx.x & 7;
    const int bbase = slice * 16;

    const float* Wbp = dir ? Wb2 : Wb1;
    const float* bbp = dir ? bb2 : bb1;
    const float* W1p = dir ? W12 : W11;
    const float* b1p = dir ? b12 : b11;
    const float* wtp = dir ? wt2 : wt1;
    const float* Ap  = dir ? A2  : A1;

    const int laneN  = lane & 15;
    const int laneHi = lane >> 4;
    const int kHalf  = laneHi << 4;      // B-frag K base within a 32-block

    // ---- Pin weight B-fragments in VGPRs for the whole scan ----------------
    // GEMM1: Wb is [512,128]; this wave owns output columns [wave*16, +16).
    v16h wbf[16];
#pragma unroll
    for (int kk = 0; kk < 16; ++kk) {
#pragma unroll
        for (int j = 0; j < 16; ++j) {
            int k = kk * 32 + kHalf + j;
            wbf[kk][j] = (_Float16)Wbp[k * CFC_BB + wave * 16 + laneN];
        }
    }
    // GEMM2: W1 is [128,256]; this wave owns columns [(wave*2+s)*16, +16).
    v16h w1f[2][4];
#pragma unroll
    for (int s = 0; s < 2; ++s) {
#pragma unroll
        for (int kk = 0; kk < 4; ++kk) {
#pragma unroll
            for (int j = 0; j < 16; ++j) {
                int k = kk * 32 + kHalf + j;
                w1f[s][kk][j] = (_Float16)W1p[k * CFC_H + (wave * 2 + s) * 16 + laneN];
            }
        }
    }

    // Biases folded into epilogues: exp arg = 1.332*acc + 1.332*bb.
    const float bb_pre = 1.332f * bbp[wave * 16 + laneN];
    float cA[2], cWt[2], cB1[2];
#pragma unroll
    for (int s = 0; s < 2; ++s) {
        int n = (wave * 2 + s) * 16 + laneN;
        cA[s]  = Ap[n];
        cWt[s] = fabsf(wtp[n]);
        cB1[s] = b1p[n];
    }

    // ---- Init hidden state to zero, preload x[0] ---------------------------
    for (int i = tid; i < 16 * SX_STRIDE; i += 256)
        (&sH[0][0])[i] = (_Float16)0.0f;

    const int mr = tid >> 4;             // 0..15: batch row within slice
    const int cc = (tid & 15) * 16;      // 16 consecutive input features
    {
        int tin0 = dir ? (CFC_T - 1) : 0;
        const float* xp = x + ((size_t)(bbase + mr) * CFC_T + tin0) * CFC_D + cc;
#pragma unroll
        for (int j = 0; j < 16; ++j) sX[0][mr][cc + j] = (_Float16)xp[j];
    }
    __syncthreads();

    // ---- The scan ----------------------------------------------------------
    for (int t = 0; t < CFC_T; ++t) {
        const int buf = t & 1;
        const bool havenext = (t + 1 < CFC_T);

        // Prefetch x_{t+1} into registers (overlaps with the WMMA chain).
        float xf[16];
        if (havenext) {
            int tin = dir ? (CFC_T - 2 - t) : (t + 1);
            const float* xp = x + ((size_t)(bbase + mr) * CFC_T + tin) * CFC_D + cc;
#pragma unroll
            for (int j = 0; j < 16; ++j) xf[j] = xp[j];
        }

        // GEMM1: z_pre[16, 16@wave] = [x_t | h] @ Wb   (K = 512, bias in epilogue)
        v8f acc1 = {};                   // zero-init -> first WMMA uses SRC2 = 0
#pragma unroll
        for (int kk = 0; kk < 8; ++kk) {
            v16h a = load_afrag(&sX[buf][0][0] + kk * 32, SX_STRIDE, lane);
            wmma_acc(a, wbf[kk], acc1);
        }
#pragma unroll
        for (int kk = 0; kk < 8; ++kk) {
            v16h a = load_afrag(&sH[0][0] + kk * 32, SX_STRIDE, lane);
            wmma_acc(a, wbf[8 + kk], acc1);
        }

        // lecun_tanh(acc + bb), publish z tile to LDS (reg r -> row r + 8*hi).
        // tanh(y) = 1 - 2/(e^{2y}+1), so exp arg = 1.332*acc + 1.332*bb (one FMA).
        const int nz = wave * 16 + laneN;
#pragma unroll
        for (int r = 0; r < 8; ++r) {
            float e  = __expf(fmaf(1.332f, acc1[r], bb_pre));
            float zv = 1.7159f * (1.0f - __fdividef(2.0f, e + 1.0f));
            sZ[r + 8 * laneHi][nz] = (_Float16)zv;
        }
        __syncthreads();

        // GEMM2 (K = 128): load the 4 shared A-fragments once, feed two
        // independent accumulator chains (column tiles s=0 and s=1).
        v16h a2[4];
#pragma unroll
        for (int kk = 0; kk < 4; ++kk)
            a2[kk] = load_afrag(&sZ[0][0] + kk * 32, SZ_STRIDE, lane);

        v8f acc2[2] = {{}, {}};
#pragma unroll
        for (int kk = 0; kk < 4; ++kk) {
            wmma_acc(a2[kk], w1f[0][kk], acc2[0]);
            wmma_acc(a2[kk], w1f[1][kk], acc2[1]);
        }

        // CfC elementwise update + output store + hidden handoff.
#pragma unroll
        for (int s = 0; s < 2; ++s) {
            const int n = (wave * 2 + s) * 16 + laneN;
#pragma unroll
            for (int r = 0; r < 8; ++r) {
                float ff1 = acc2[s][r] + cB1[s];
                float hn  = cA[s] - cA[s] * __expf(-(cWt[s] + fabsf(ff1))) * ff1;
                int m = r + 8 * laneHi;
                // y[b, t, dir*H + n]; reversed direction is NOT flipped back.
                out[(((size_t)(bbase + m)) * CFC_T + t) * (2 * CFC_H) + dir * CFC_H + n] = hn;
                sH[m][n] = (_Float16)hn;
            }
        }

        // Commit prefetched x_{t+1} to the other buffer before closing barrier.
        if (havenext) {
#pragma unroll
            for (int j = 0; j < 16; ++j) sX[buf ^ 1][mr][cc + j] = (_Float16)xf[j];
        }
        __syncthreads();
    }
}

extern "C" void kernel_launch(void* const* d_in, const int* in_sizes, int n_in,
                              void* d_out, int out_size, void* d_ws, size_t ws_size,
                              hipStream_t stream) {
    const float* x   = (const float*)d_in[0];
    const float* Wb1 = (const float*)d_in[1];
    const float* bb1 = (const float*)d_in[2];
    const float* W11 = (const float*)d_in[3];
    const float* b11 = (const float*)d_in[4];
    const float* wt1 = (const float*)d_in[5];
    const float* A1  = (const float*)d_in[6];
    const float* Wb2 = (const float*)d_in[7];
    const float* bb2 = (const float*)d_in[8];
    const float* W12 = (const float*)d_in[9];
    const float* b12 = (const float*)d_in[10];
    const float* wt2 = (const float*)d_in[11];
    const float* A2  = (const float*)d_in[12];
    float* out = (float*)d_out;

    (void)in_sizes; (void)n_in; (void)out_size; (void)d_ws; (void)ws_size;

    cfc_scan_kernel<<<dim3(16), dim3(256), 0, stream>>>(
        x, Wb1, bb1, W11, b11, wt1, A1, Wb2, bb2, W12, b12, wt2, A2, out);
}
